// DyChannel_24051816858031
// MI455X (gfx1250) — compile-verified, matrix operand order checked by reading, not run on hardware
//
#include <hip/hip_runtime.h>
#include <hip/hip_bf16.h>

typedef __attribute__((ext_vector_type(16))) _Float16 v16h;
typedef __attribute__((ext_vector_type(8)))  float    v8f;

union V8F { v8f v; float e[8]; };

#define B_SZ  16
#define CIN   256
#define HH    64
#define WW    64
#define COUT  256
#define NEXP  4
#define REDC  16

// ---------------------------------------------------------------------------
// Kernel 1: routing network.  One block per sample.
// pooled = mean(x, HW); r = relu(pooled @ w1^T + b1); rw = sigmoid(r @ w2^T + b2)
// ---------------------------------------------------------------------------
__global__ __launch_bounds__(256) void routing_kernel(
    const float* __restrict__ x, const float* __restrict__ w1,
    const float* __restrict__ b1, const float* __restrict__ w2,
    const float* __restrict__ b2, float* __restrict__ rw)
{
    __shared__ float pooled[CIN];
    __shared__ float rbuf[REDC];
    const int b    = blockIdx.x;
    const int t    = threadIdx.x;
    const int lane = t & 31;
    const int wave = t >> 5;

    for (int c = wave * 32; c < wave * 32 + 32; ++c) {
        const float* p = x + ((size_t)(b * CIN + c)) * (HH * WW);
        float s = 0.0f;
        for (int i = lane; i < HH * WW; i += 32) s += p[i];
        #pragma unroll
        for (int off = 16; off > 0; off >>= 1) s += __shfl_down(s, off, 32);
        if (lane == 0) pooled[c] = s * (1.0f / (HH * WW));
    }
    __syncthreads();

    if (t < REDC) {
        float s = b1[t];
        for (int c = 0; c < CIN; ++c) s += pooled[c] * w1[t * CIN + c];
        rbuf[t] = fmaxf(s, 0.0f);
    }
    __syncthreads();

    for (int idx = t; idx < NEXP * CIN; idx += 256) {
        float s = b2[idx];
        #pragma unroll
        for (int j = 0; j < REDC; ++j) s += rbuf[j] * w2[idx * REDC + j];
        rw[b * (NEXP * CIN) + idx] = 1.0f / (1.0f + __expf(-s));
    }
}

// ---------------------------------------------------------------------------
// Kernel 2: build per-sample combined f16 weights directly in WMMA A-fragment
// order.  Logical value: wcomb(b,ij,c,o) = sum_k rw[b,k,c] * weight[k,o,c,ij].
// Storage: fragment id f = ((b*9+ij)*8 + kc)*16 + mt  (kc = c>>5, mt = o>>4),
// inside a fragment: [lane][elem] with
//   lane = (o & 15) | (((c>>3)&1) << 4)
//   elem = (c & 7)  | (((c>>4)&1) << 3)      (ISA 16-bit A 16x32 layout)
// Thread mapping is the inverse so writes are fully contiguous/coalesced.
// ---------------------------------------------------------------------------
__global__ __launch_bounds__(256) void combine_kernel(
    const float* __restrict__ rw, const float* __restrict__ weight,
    _Float16* __restrict__ wcomb)
{
    const int gid  = blockIdx.x * 256 + threadIdx.x;    // < 16*9*8*16*32*16
    const int elem = gid & 15;
    const int lane = (gid >> 4) & 31;
    const int mt   = (gid >> 9) & 15;
    const int kc   = (gid >> 13) & 7;
    const int rest = gid >> 16;                         // b*9 + ij
    const int ij   = rest % 9;
    const int b    = rest / 9;

    const int c = (elem & 7) | (((lane >> 4) & 1) << 3) |
                  (((elem >> 3) & 1) << 4) | (kc << 5);
    const int o = (lane & 15) | (mt << 4);

    float s = 0.0f;
    #pragma unroll
    for (int k = 0; k < NEXP; ++k) {
        float r = rw[(b * NEXP + k) * CIN + c];
        float w = weight[(((size_t)(k * COUT + o) * CIN + c) * 9) + ij];
        s += r * w;
    }
    wcomb[gid] = (_Float16)s;
}

// ---------------------------------------------------------------------------
// Kernel 3: implicit-GEMM 3x3 conv via v_wmma_f32_16x16x32_f16.
// Block = (h row, sample b): out[b,0:256,h,0:64]  (M=256, N=64).
// Static nested loops: ij outer (block-uniform skip), kc inner FULLY UNROLLED
// with double-buffered LDS B fragments and register-double-buffered A
// fragments -> one barrier per K-step, loads hidden under WMMAs, all offsets
// compile-time constants.
// ---------------------------------------------------------------------------
__global__ __launch_bounds__(256) void conv_wmma_kernel(
    const float* __restrict__ x, const _Float16* __restrict__ wcomb,
    float* __restrict__ out)
{
    // double-buffered B fragments: [buf][ni(4)][lane(32)][elem(16)] = 2 x 4 KB
    __shared__ __align__(32) _Float16 Bfrag[2][4 * 32 * 16];

    const int h    = blockIdx.x;
    const int b    = blockIdx.y;
    const int t    = threadIdx.x;
    const int lane = t & 31;
    const int wave = t >> 5;

    v8f zero = {};
    v8f acc[2][4];
    #pragma unroll
    for (int mi = 0; mi < 2; ++mi)
        #pragma unroll
        for (int ni = 0; ni < 4; ++ni) acc[mi][ni] = zero;

    // B-staging constants: this thread handles channel-in-chunk kci, 8 pixels
    const int kci     = t >> 3;
    const int w0      = (t & 7) * 8;
    const int lane_hi = ((kci >> 3) & 1) << 4;
    const int belem   = (kci & 7) | (((kci >> 4) & 1) << 3);

    auto stageB = [&](int hs, int dj, int kc, int buf) {
        const float* xrow =
            x + ((size_t)(b * CIN + kc * 32 + kci) * HH + hs) * WW;
        #pragma unroll
        for (int e = 0; e < 8; ++e) {
            int n    = w0 + e;
            int wsrc = n + dj;
            float v  = (wsrc >= 0 && wsrc < WW) ? xrow[wsrc] : 0.0f;
            int ni   = n >> 4;
            int bl   = (n & 15) | lane_hi;
            Bfrag[buf][((ni * 32 + bl) << 4) + belem] = (_Float16)v;
        }
    };

    const int aoff0 = (wave * 64 + lane) << 4;          // mt = wave*2 + 0
    const int aoff1 = (wave * 64 + 32 + lane) << 4;     // mt = wave*2 + 1

    for (int ij = 0; ij < 9; ++ij) {
        const int di = ij / 3 - 1;
        const int dj = ij % 3 - 1;
        const int hs = h + di;
        if (hs < 0 || hs >= HH) continue;               // block-uniform

        // A chunks for this ij are contiguous: chunk kc at halves kc<<13
        const _Float16* abase =
            wcomb + ((size_t)((b * 9 + ij) * 8) << 13);

        // ---- per-ij prologue: stage kc=0, load its A fragments
        stageB(hs, dj, 0, 0);
        v16h a0 = *(const v16h*)(abase + aoff0);
        v16h a1 = *(const v16h*)(abase + aoff1);
        __builtin_prefetch(abase + ((size_t)8 << 13), 0, 0); // next ij region
        __syncthreads();

        #pragma unroll
        for (int kc = 0; kc < 8; ++kc) {
            const int buf = kc & 1;

            // stage next step + next A fragments under current WMMAs
            v16h an0 = a0, an1 = a1;
            if (kc + 1 < 8) {
                stageB(hs, dj, kc + 1, buf ^ 1);
                const _Float16* abn = abase + ((size_t)(kc + 1) << 13);
                an0 = *(const v16h*)(abn + aoff0);
                an1 = *(const v16h*)(abn + aoff1);
            }

            v16h bf[4];
            #pragma unroll
            for (int ni = 0; ni < 4; ++ni)
                bf[ni] = *(const v16h*)(&Bfrag[buf][(ni * 32 + lane) << 4]);

            #pragma unroll
            for (int ni = 0; ni < 4; ++ni) {
                acc[0][ni] = __builtin_amdgcn_wmma_f32_16x16x32_f16(
                    false, a0, false, bf[ni], (short)0, acc[0][ni], false, false);
                acc[1][ni] = __builtin_amdgcn_wmma_f32_16x16x32_f16(
                    false, a1, false, bf[ni], (short)0, acc[1][ni], false, false);
            }
            a0 = an0;
            a1 = an1;
            __syncthreads();
        }
    }

    // ---- epilogue: ISA C/D layout -> out[b][o][h][w]
    const int nLane = lane & 15;
    const int mOff  = (lane >> 4) * 8;
    #pragma unroll
    for (int mi = 0; mi < 2; ++mi) {
        #pragma unroll
        for (int ni = 0; ni < 4; ++ni) {
            V8F r; r.v = acc[mi][ni];
            #pragma unroll
            for (int q = 0; q < 8; ++q) {
                int o    = wave * 32 + mi * 16 + mOff + q;
                int wcol = ni * 16 + nLane;
                out[((size_t)(b * COUT + o) * HH + h) * WW + wcol] = r.e[q];
            }
        }
    }
}

// ---------------------------------------------------------------------------
extern "C" void kernel_launch(void* const* d_in, const int* in_sizes, int n_in,
                              void* d_out, int out_size, void* d_ws, size_t ws_size,
                              hipStream_t stream) {
    const float* x      = (const float*)d_in[0];
    const float* w1     = (const float*)d_in[1];
    const float* b1     = (const float*)d_in[2];
    const float* w2     = (const float*)d_in[3];
    const float* b2     = (const float*)d_in[4];
    const float* weight = (const float*)d_in[5];
    float* out = (float*)d_out;

    // workspace layout: rw (16*1024 f32) | wcomb fragments (16*9*256*256 f16)
    float*    rw    = (float*)d_ws;
    _Float16* wcomb = (_Float16*)((char*)d_ws + (size_t)B_SZ * NEXP * CIN * sizeof(float));

    routing_kernel<<<B_SZ, 256, 0, stream>>>(x, w1, b1, w2, b2, rw);

    const int combine_total = B_SZ * 9 * CIN * COUT;            // 9,437,184
    combine_kernel<<<combine_total / 256, 256, 0, stream>>>(rw, weight, wcomb);

    dim3 grid(HH, B_SZ);
    conv_wmma_kernel<<<grid, 256, 0, stream>>>(x, wcomb, out);
}
